// TemporalTransportGNN_18219251270346
// MI455X (gfx1250) — matrix operand order repaired and checked by reading, not compiled, not run on hardware
//
#include <hip/hip_runtime.h>

// ---------------------------------------------------------------------------
// CDNA5 (gfx1250) TemporalTransportGNN
//   - all dense layers via v_wmma_f32_16x16x32_bf16 (wave32, 8 waves/block)
//   - graph aggregation via f32 global atomics
// ---------------------------------------------------------------------------

typedef __attribute__((ext_vector_type(16))) __bf16 v16bf;
typedef __attribute__((ext_vector_type(8)))  __bf16 v8bf;
typedef __attribute__((ext_vector_type(8)))  float  v8f;

static __device__ __forceinline__ v16bf join8(v8bf lo, v8bf hi) {
    return __builtin_shufflevector(lo, hi, 0,1,2,3,4,5,6,7,8,9,10,11,12,13,14,15);
}

// ---------------------------------------------------------------------------
// Generic bf16 WMMA GEMM:  C[M,NN] = op(A[M,K] @ Bw[K,NN] + bias)
//   A row-major bf16, Bw row-major bf16 (K x NN), C row-major (ldc, colOff)
//   Block = 256 threads = 8 waves; tile = 128 rows x BN cols; M % 128 == 0.
//   Wave w owns rows [blk*128 + w*16, +16); BN/16 16x16 f32 accumulators.
// ---------------------------------------------------------------------------
template<int BN, bool RELU, bool OUT_BF16, bool BIAS>
__global__ __launch_bounds__(256) void gemm_wmma(
        const __bf16* __restrict__ A, const __bf16* __restrict__ Bw,
        const float* __restrict__ bias, void* __restrict__ C,
        int K, int NN, int ldc, int colOff) {
    static_assert(BN == 64 || BN == 128, "BN must be 64 or 128");
    constexpr int NT = BN / 16;

    const int wave = threadIdx.x >> 5;
    const int lane = threadIdx.x & 31;
    const int m0   = blockIdx.x * 128 + wave * 16;
    const int n0   = blockIdx.y * BN;

    const int mrow = lane & 15;
    const int half = lane >> 4;

    v8f acc[NT];
#pragma unroll
    for (int t = 0; t < NT; ++t) acc[t] = v8f{0.f,0.f,0.f,0.f,0.f,0.f,0.f,0.f};

    // A fragment base for this lane: row m0+mrow, K offset half*8 (+16 for hi run)
    const __bf16* arow = A + (size_t)(m0 + mrow) * K + half * 8;

    for (int k0 = 0; k0 < K; k0 += 32) {
        v8bf lo = *(const v8bf*)(arow + k0);        // K = k0 + half*8 + [0..7]
        v8bf hi = *(const v8bf*)(arow + k0 + 16);   // K = k0 + 16 + half*8 + [0..7]
        v16bf afrag = join8(lo, hi);

        // B fragment: lane holds col N = n0+t*16+(lane&15), K = k0 + half*8 + [0..7], k0 + 16 + half*8 + [0..7]
        const __bf16* bcol = Bw + (size_t)(k0 + half * 8) * NN + n0 + mrow;
#pragma unroll
        for (int t = 0; t < NT; ++t) {
            v16bf bfrag;
#pragma unroll
            for (int s = 0; s < 16; ++s)
                bfrag[s] = bcol[(size_t)(s + (s >= 8 ? 8 : 0)) * NN + t * 16];
            acc[t] = __builtin_amdgcn_wmma_f32_16x16x32_bf16(
                         false, afrag, false, bfrag,
                         (short)0, acc[t], false, false);
        }
    }

    // Epilogue: C layout — VGPR v -> M = v + 8*half, N = lane&15
#pragma unroll
    for (int t = 0; t < NT; ++t) {
        const int cn = n0 + t * 16 + (lane & 15);
        float badd = BIAS ? bias[cn] : 0.f;
#pragma unroll
        for (int v = 0; v < 8; ++v) {
            const int cm = m0 + v + 8 * half;
            float val = acc[t][v] + badd;
            if (RELU) val = fmaxf(val, 0.f);
            const size_t off = (size_t)cm * ldc + colOff + cn;
            if (OUT_BF16) ((__bf16*)C)[off] = (__bf16)val;
            else          ((float*)C)[off]  = val;
        }
    }
}

// ---------------------------------------------------------------------------
// Utility / graph kernels
// ---------------------------------------------------------------------------
__global__ void zero_f32(float* __restrict__ p, long long n) {
    long long i = (long long)blockIdx.x * blockDim.x + threadIdx.x;
    if (i < n) p[i] = 0.f;
}

// f32 -> bf16 with zero row padding (row-major [rows,cols] -> [padRows,cols])
__global__ void cvt_pad_bf16(const float* __restrict__ src, __bf16* __restrict__ dst,
                             int rows, int cols, long long total) {
    long long i = (long long)blockIdx.x * blockDim.x + threadIdx.x;
    if (i >= total) return;
    int r = (int)(i / cols);
    dst[i] = (__bf16)((r < rows) ? src[i] : 0.f);
}

__global__ void deg_accum(const int* __restrict__ dst, float* __restrict__ deg, int E) {
    int e = blockIdx.x * blockDim.x + threadIdx.x;
    if (e < E) atomicAdd(&deg[dst[e]], 1.f);
}

__global__ void deg_to_dis(const float* __restrict__ deg, float* __restrict__ dis, int N) {
    int i = blockIdx.x * blockDim.x + threadIdx.x;
    if (i < N) dis[i] = rsqrtf(deg[i] + 1.f);   // +1 = self loop
}

__global__ void edge_norm(const int* __restrict__ src, const int* __restrict__ dst,
                          const float* __restrict__ dis, float* __restrict__ nrm, int E) {
    int e = blockIdx.x * blockDim.x + threadIdx.x;
    if (e < E) nrm[e] = dis[src[e]] * dis[dst[e]];
}

// hagg[i,:] = hlin[i,:] * dis[i]^2   (self-loop term, also initializes hagg)
__global__ void self_loop_init(const float* __restrict__ hlin, const float* __restrict__ dis,
                               float* __restrict__ hagg, int N) {
    long long i = (long long)blockIdx.x * blockDim.x + threadIdx.x;
    if (i >= (long long)N * 128) return;
    float d = dis[(int)(i >> 7)];
    hagg[i] = hlin[i] * d * d;
}

// per edge: hagg[dst] += hlin[src] * norm   (32 threads/edge, float4 each)
__global__ __launch_bounds__(256) void edge_scatter(
        const float* __restrict__ hlin, const int* __restrict__ src,
        const int* __restrict__ dst, const float* __restrict__ nrm,
        float* __restrict__ hagg, int E) {
    long long tid = (long long)blockIdx.x * blockDim.x + threadIdx.x;
    if (tid >= (long long)E * 32) return;
    int e = (int)(tid >> 5);
    int q = (int)(tid & 31);
    const float4 v = *(const float4*)(hlin + (size_t)src[e] * 128 + q * 4);
    float w = nrm[e];
    float* p = hagg + (size_t)dst[e] * 128 + q * 4;
    atomicAdd(p + 0, v.x * w);
    atomicAdd(p + 1, v.y * w);
    atomicAdd(p + 2, v.z * w);
    atomicAdd(p + 3, v.w * w);
}

// hin[i,c] = bf16(relu(hagg[i,c] + b[c])), zero padding rows
__global__ void bias_relu_cvt(const float* __restrict__ hagg, const float* __restrict__ b,
                              __bf16* __restrict__ hin, int N, long long total) {
    long long i = (long long)blockIdx.x * blockDim.x + threadIdx.x;
    if (i >= total) return;
    int r = (int)(i >> 7), c = (int)(i & 127);
    float v = 0.f;
    if (r < N) v = fmaxf(hagg[i] + b[c], 0.f);
    hin[i] = (__bf16)v;
}

// T[B,128] bf16 = [day_emb | time_emb | dow_emb | 0 0]
__global__ void build_temporal(const float* __restrict__ day_emb, const float* __restrict__ time_emb,
                               const float* __restrict__ dow_emb, const int* __restrict__ day_id,
                               const int* __restrict__ time_id, const int* __restrict__ dow_id,
                               __bf16* __restrict__ T, int B) {
    long long idx = (long long)blockIdx.x * blockDim.x + threadIdx.x;
    if (idx >= (long long)B * 128) return;
    int i = (int)(idx >> 7), c = (int)(idx & 127);
    float v = 0.f;
    if (c < 42)       v = day_emb[day_id[i] * 42 + c];
    else if (c < 84)  v = time_emb[time_id[i] * 42 + (c - 42)];
    else if (c < 126) v = dow_emb[dow_id[i] * 42 + (c - 84)];
    T[idx] = (__bf16)v;
}

// z cols [0,256): gathered h3[origin]|h3[dest]; cols [384,448): mode_emb
__global__ void build_route_mode(const __bf16* __restrict__ h3, const int* __restrict__ origin,
                                 const int* __restrict__ dest, const float* __restrict__ mode_emb,
                                 const int* __restrict__ mode_id, __bf16* __restrict__ z, int B) {
    long long idx = (long long)blockIdx.x * blockDim.x + threadIdx.x;
    if (idx >= (long long)B * 320) return;
    int i = (int)(idx / 320), c = (int)(idx - (long long)i * 320);
    __bf16* zr = z + (size_t)i * 448;
    if (c < 128)      zr[c] = h3[(size_t)origin[i] * 128 + c];
    else if (c < 256) zr[c] = h3[(size_t)dest[i]   * 128 + (c - 128)];
    else              zr[384 + (c - 256)] = (__bf16)mode_emb[mode_id[i] * 64 + (c - 256)];
}

// score = sigmoid(a3 @ Wh4 + bh4), output f32
__global__ void head_final(const __bf16* __restrict__ a3, const float* __restrict__ Wh4,
                           const float* __restrict__ bh4, float* __restrict__ out, int B) {
    int i = blockIdx.x * blockDim.x + threadIdx.x;
    if (i >= B) return;
    const __bf16* r = a3 + (size_t)i * 64;
    float s = 0.f;
#pragma unroll
    for (int j = 0; j < 64; ++j) s += (float)r[j] * Wh4[j];
    s += bh4[0];
    out[i] = 1.f / (1.f + __expf(-s));
}

// ---------------------------------------------------------------------------
extern "C" void kernel_launch(void* const* d_in, const int* in_sizes, int n_in,
                              void* d_out, int out_size, void* d_ws, size_t ws_size,
                              hipStream_t stream) {
    const int N = in_sizes[0] / 128;      // 100000
    const int E = in_sizes[1];            // 1600000
    const int B = in_sizes[3];            // 16384
    const int Npad = ((N + 127) / 128) * 128;

    const float* x        = (const float*)d_in[0];
    const int*   srcI     = (const int*)d_in[1];
    const int*   dstI     = (const int*)d_in[2];
    const int*   origin   = (const int*)d_in[3];
    const int*   dest     = (const int*)d_in[4];
    const int*   day_id   = (const int*)d_in[5];
    const int*   time_id  = (const int*)d_in[6];
    const int*   dow_id   = (const int*)d_in[7];
    const int*   mode_id  = (const int*)d_in[8];
    const float* Wgcn[3]  = {(const float*)d_in[9], (const float*)d_in[11], (const float*)d_in[13]};
    const float* bgcn[3]  = {(const float*)d_in[10], (const float*)d_in[12], (const float*)d_in[14]};
    const float* day_emb  = (const float*)d_in[15];
    const float* time_emb = (const float*)d_in[16];
    const float* dow_emb  = (const float*)d_in[17];
    const float* Wf       = (const float*)d_in[18];
    const float* bf_      = (const float*)d_in[19];
    const float* mode_emb = (const float*)d_in[20];
    const float* Wh1      = (const float*)d_in[21];
    const float* bh1      = (const float*)d_in[22];
    const float* Wh2      = (const float*)d_in[23];
    const float* bh2      = (const float*)d_in[24];
    const float* Wh3      = (const float*)d_in[25];
    const float* bh3      = (const float*)d_in[26];
    const float* Wh4      = (const float*)d_in[27];
    const float* bh4      = (const float*)d_in[28];
    float* out = (float*)d_out;

    // ---- workspace carve-out (256B aligned) ----
    char* wp = (char*)d_ws;
    auto alloc = [&](size_t bytes) -> void* {
        void* p = wp; wp += (bytes + 255) & ~(size_t)255; return p;
    };
    float*  deg   = (float*)alloc((size_t)N * 4);
    float*  dis   = (float*)alloc((size_t)N * 4);
    float*  nrm   = (float*)alloc((size_t)E * 4);
    __bf16* hin   = (__bf16*)alloc((size_t)Npad * 128 * 2);
    float*  hlin  = (float*)alloc((size_t)Npad * 128 * 4);
    float*  hagg  = (float*)alloc((size_t)Npad * 128 * 4);
    __bf16* wg_bf[3];
    for (int l = 0; l < 3; ++l) wg_bf[l] = (__bf16*)alloc(128 * 128 * 2);
    __bf16* wf_bf  = (__bf16*)alloc(128 * 128 * 2);        // Wf padded 126->128 rows
    __bf16* wh1_bf = (__bf16*)alloc(448 * 256 * 2);
    __bf16* wh2_bf = (__bf16*)alloc(256 * 128 * 2);
    __bf16* wh3_bf = (__bf16*)alloc(128 * 64 * 2);
    __bf16* Tbuf   = (__bf16*)alloc((size_t)B * 128 * 2);
    __bf16* zbuf   = (__bf16*)alloc((size_t)B * 448 * 2);
    __bf16* a1     = (__bf16*)alloc((size_t)B * 256 * 2);
    __bf16* a2     = (__bf16*)alloc((size_t)B * 128 * 2);
    __bf16* a3     = (__bf16*)alloc((size_t)B * 64 * 2);

    auto blocks = [](long long n) { return (int)((n + 255) / 256); };

    // ---- degree / normalization ----
    zero_f32<<<blocks(N), 256, 0, stream>>>(deg, N);
    deg_accum<<<blocks(E), 256, 0, stream>>>(dstI, deg, E);
    deg_to_dis<<<blocks(N), 256, 0, stream>>>(deg, dis, N);
    edge_norm<<<blocks(E), 256, 0, stream>>>(srcI, dstI, dis, nrm, E);

    // ---- weight conversions to bf16 ----
    for (int l = 0; l < 3; ++l)
        cvt_pad_bf16<<<blocks(128 * 128), 256, 0, stream>>>(Wgcn[l], wg_bf[l], 128, 128, 128 * 128);
    cvt_pad_bf16<<<blocks(128 * 128), 256, 0, stream>>>(Wf, wf_bf, 126, 128, 128 * 128);
    cvt_pad_bf16<<<blocks(448 * 256), 256, 0, stream>>>(Wh1, wh1_bf, 448, 256, 448 * 256);
    cvt_pad_bf16<<<blocks(256 * 128), 256, 0, stream>>>(Wh2, wh2_bf, 256, 128, 256 * 128);
    cvt_pad_bf16<<<blocks(128 * 64), 256, 0, stream>>>(Wh3, wh3_bf, 128, 64, 128 * 64);

    // ---- node features -> bf16 (padded rows) ----
    const long long nodeTot = (long long)Npad * 128;
    cvt_pad_bf16<<<blocks(nodeTot), 256, 0, stream>>>(x, hin, N, 128, nodeTot);

    // ---- 3 GCN layers ----
    for (int l = 0; l < 3; ++l) {
        // hlin = hin @ Wl     (WMMA bf16 -> f32)
        gemm_wmma<128, false, false, false>
            <<<dim3(Npad / 128, 1), 256, 0, stream>>>(hin, wg_bf[l], nullptr, hlin, 128, 128, 128, 0);
        // hagg = hlin * dis^2 (self loop); then scatter edges
        self_loop_init<<<blocks((long long)N * 128), 256, 0, stream>>>(hlin, dis, hagg, N);
        edge_scatter<<<blocks((long long)E * 32), 256, 0, stream>>>(hlin, srcI, dstI, nrm, hagg, E);
        // hin = bf16(relu(hagg + b))
        bias_relu_cvt<<<blocks(nodeTot), 256, 0, stream>>>(hagg, bgcn[l], hin, N, nodeTot);
    }

    // ---- temporal branch: T[B,128] -> relu(T@Wf+bf) into z cols [256,384) ----
    build_temporal<<<blocks((long long)B * 128), 256, 0, stream>>>(
        day_emb, time_emb, dow_emb, day_id, time_id, dow_id, Tbuf, B);
    gemm_wmma<128, true, true, true>
        <<<dim3(B / 128, 1), 256, 0, stream>>>(Tbuf, wf_bf, bf_, zbuf, 128, 128, 448, 256);

    // ---- route gather + mode embedding into z ----
    build_route_mode<<<blocks((long long)B * 320), 256, 0, stream>>>(
        hin, origin, dest, mode_emb, mode_id, zbuf, B);

    // ---- prediction head ----
    gemm_wmma<128, true, true, true>
        <<<dim3(B / 128, 2), 256, 0, stream>>>(zbuf, wh1_bf, bh1, a1, 448, 256, 256, 0);
    gemm_wmma<128, true, true, true>
        <<<dim3(B / 128, 1), 256, 0, stream>>>(a1, wh2_bf, bh2, a2, 256, 128, 128, 0);
    gemm_wmma<64, true, true, true>
        <<<dim3(B / 128, 1), 256, 0, stream>>>(a2, wh3_bf, bh3, a3, 128, 64, 64, 0);
    head_final<<<blocks(B), 256, 0, stream>>>(a3, Wh4, bh4, out, B);
}
